// DualTransformerBlock_63702954934875
// MI455X (gfx1250) — compile-verified, matrix operand order checked by reading, not run on hardware
//
#include <hip/hip_runtime.h>
#include <hip/hip_bf16.h>

#define DEV __device__ __forceinline__

typedef __attribute__((ext_vector_type(16))) __bf16 v16bf;
typedef __attribute__((ext_vector_type(8)))  float  v8f;

union Frag { v16bf v; unsigned u[8]; };

DEV float  tof(__bf16 h){ return (float)h; }
DEV __bf16 tobf(float f){ return (__bf16)f; }

// Load a 16x32 (or 32x16-transposed) bf16 WMMA fragment.  `p` is this lane's
// row pointer already offset by kb = ((lane>>4)<<3).  Per ISA 7.12.2:
// VGPR v holds K pair k0 = (v&4 ? 16:0) + kb + 2*(v&3); pairs are contiguous
// in memory -> one 32-bit load each (compiler vectorizes to b128).
DEV void load_frag(Frag& f, const __bf16* p){
  #pragma unroll
  for (int i = 0; i < 8; ++i) {
    int off = ((i & 4) << 2) + ((i & 3) << 1);
    unsigned t; __builtin_memcpy(&t, p + off, 4);
    f.u[i] = t;
  }
}

DEV v8f wmma_bf16(const Frag& a, const Frag& b, v8f c){
  return __builtin_amdgcn_wmma_f32_16x16x32_bf16(
      /*neg_a=*/false, a.v, /*neg_b=*/false, b.v,
      /*c_mod=*/(short)0, c, /*reuse_a=*/false, /*reuse_b=*/false);
}

// Async copy of one 16B chunk: LDS[lds_off] = MEM[saddr + goff]  (ASYNCcnt)
DEV void async_b128(unsigned lds_off, int goff, const void* gbase){
  asm volatile("global_load_async_to_lds_b128 %0, %1, %2"
               :: "v"(lds_off), "v"(goff), "s"(gbase) : "memory");
}
DEV void wait_async0(){ asm volatile("s_wait_asynccnt 0x0" ::: "memory"); }

// ---------------- problem constants ----------------
constexpr int Bc   = 32;
constexpr int Hh   = 56;
constexpr int Ww   = 56;
constexpr int Nn   = Hh * Ww;                 // 3136
constexpr int Mrows= Bc * Nn;                 // 100352
constexpr int H2   = 4;                       // heads per branch
constexpr int N1   = 14 * 14;                 // 196 SR tokens per image
constexpr int P1   = Bc * N1;                 // 6272
constexpr int KV1R = P1 + 32;                 // padded rows (OOB-safe tile reads)
constexpr float SCALE = 0.17677669529663687f; // 32^-0.5

// ---------------- conversion / prep ----------------
__global__ void cvt_bf16_kernel(const float* __restrict__ in, __bf16* __restrict__ out, int n){
  int i = blockIdx.x * 256 + threadIdx.x;
  if (i < n) out[i] = tobf(in[i]);
}

__global__ void prep_weights_kernel(const float* lepe_lin_w, const float* q1_w,
                                    const float* q2_w, const float* kv2_w,
                                    const float* kv1_w, const float* proj_w,
                                    const float* sr_w,
                                    __bf16* wcat, __bf16* wkv1t, __bf16* wprojt, __bf16* wsrt){
  int i = blockIdx.x * 256 + threadIdx.x;     // 1376256 total
  if (i < 196608) {                           // Wcat_t [768][256]
    int n = i >> 8, k = i & 255; float v;
    if      (n < 256) v = lepe_lin_w[k*256 + n];
    else if (n < 384) v = q1_w[k*128 + (n-256)];
    else if (n < 512) v = q2_w[k*128 + (n-384)];
    else              v = kv2_w[k*256 + (n-512)];
    wcat[n*256 + k] = tobf(v);
  } else if (i < 262144) {                    // Wkv1_t [256][256]
    int j = i - 196608; int n = j >> 8, k = j & 255;
    wkv1t[j] = tobf(kv1_w[k*256 + n]);
  } else if (i < 327680) {                    // Wproj_t [256][256]
    int j = i - 262144; int n = j >> 8, k = j & 255;
    wprojt[j] = tobf(proj_w[k*256 + n]);
  } else if (i < 1376256) {                   // Wsr_t [256][16][256]
    int j = i - 327680; int cout = j >> 12, tap = (j >> 8) & 15, cin = j & 255;
    wsrt[j] = tobf(sr_w[tap*65536 + cin*256 + cout]);
  }
}

// -------- fused x-projection GEMM: [100352,256] x [256,768] --------
// Block = 8 waves, one 64-wide N group staged in LDS via async copy; each
// wave: 1 M tile x 4 N tiles.  n<256: lepe_lin(+bias); 256..383: q1;
// 384..511: q2; 512..767: kv2  (all boundaries are 64-aligned).
__global__ void gemm_fused_kernel(const __bf16* __restrict__ xb, const __bf16* __restrict__ wcat,
                                  const float* __restrict__ lin_b,
                                  __bf16* lin, __bf16* q1, __bf16* q2, __bf16* kv2){
  __shared__ __bf16 bsh[64 * 256];            // 32 KB weight slab
  int gn = blockIdx.x % 12;                   // N group (64 cols)
  int bm = blockIdx.x / 12;                   // M block (8 tiles)
  int wv = threadIdx.x >> 5, lane = threadIdx.x & 31;
  int nloc = lane & 15, hi = lane >> 4, kb = hi << 3;

  {
    const __bf16* gsrc = wcat + (size_t)gn * 64 * 256;
    unsigned lbase = (unsigned)(uintptr_t)(&bsh[0]);
    for (int e = threadIdx.x; e < 2048; e += 256)
      async_b128(lbase + e * 16, e * 16, gsrc);
    wait_async0();
  }
  __syncthreads();

  int tm = bm * 8 + wv;
  const __bf16* arow = xb + (size_t)(tm*16 + nloc)*256 + kb;
  v8f acc0 = {}, acc1 = {}, acc2 = {}, acc3 = {};
  #pragma unroll
  for (int kk = 0; kk < 8; ++kk) {
    __builtin_prefetch(arow + kk*32 + 256, 0, 1);
    Frag a; load_frag(a, arow + kk*32);
    Frag b0, b1, b2, b3;
    load_frag(b0, &bsh[(0*16 + nloc)*256 + kk*32 + kb]);
    load_frag(b1, &bsh[(1*16 + nloc)*256 + kk*32 + kb]);
    load_frag(b2, &bsh[(2*16 + nloc)*256 + kk*32 + kb]);
    load_frag(b3, &bsh[(3*16 + nloc)*256 + kk*32 + kb]);
    acc0 = wmma_bf16(a, b0, acc0);
    acc1 = wmma_bf16(a, b1, acc1);
    acc2 = wmma_bf16(a, b2, acc2);
    acc3 = wmma_bf16(a, b3, acc3);
  }
  #pragma unroll
  for (int t = 0; t < 4; ++t) {
    v8f acc = (t == 0) ? acc0 : (t == 1) ? acc1 : (t == 2) ? acc2 : acc3;
    int n = gn*64 + t*16 + nloc;
    #pragma unroll
    for (int r = 0; r < 8; ++r) {
      size_t m = (size_t)tm*16 + r + (hi << 3);
      float v = acc[r];
      if      (n < 256) lin[m*256 + n]        = tobf(v + lin_b[n]);
      else if (n < 384) q1 [m*128 + (n-256)]  = tobf(v);
      else if (n < 512) q2 [m*128 + (n-384)]  = tobf(v);
      else              kv2[m*256 + (n-512)]  = tobf(v);
    }
  }
}

// -------- generic K=256 GEMM with LDS-staged B (async), bf16 or f32 out ----
// grid = (Mtiles/8) * 4;  block = 256
__global__ void gemm_k256_kernel(const __bf16* __restrict__ A, const __bf16* __restrict__ Bt,
                                 const float* __restrict__ bias,
                                 __bf16* outbf, float* outf){
  __shared__ __bf16 bsh[64 * 256];
  int gn = blockIdx.x & 3;
  int bm = blockIdx.x >> 2;
  int wv = threadIdx.x >> 5, lane = threadIdx.x & 31;
  int nloc = lane & 15, hi = lane >> 4, kb = hi << 3;

  {
    const __bf16* gsrc = Bt + (size_t)gn * 64 * 256;
    unsigned lbase = (unsigned)(uintptr_t)(&bsh[0]);
    for (int e = threadIdx.x; e < 2048; e += 256)
      async_b128(lbase + e * 16, e * 16, gsrc);
    wait_async0();
  }
  __syncthreads();

  int tm = bm * 8 + wv;
  const __bf16* arow = A + (size_t)(tm*16 + nloc)*256 + kb;
  v8f acc0 = {}, acc1 = {}, acc2 = {}, acc3 = {};
  #pragma unroll
  for (int kk = 0; kk < 8; ++kk) {
    __builtin_prefetch(arow + kk*32 + 256, 0, 1);
    Frag a; load_frag(a, arow + kk*32);
    Frag b0, b1, b2, b3;
    load_frag(b0, &bsh[(0*16 + nloc)*256 + kk*32 + kb]);
    load_frag(b1, &bsh[(1*16 + nloc)*256 + kk*32 + kb]);
    load_frag(b2, &bsh[(2*16 + nloc)*256 + kk*32 + kb]);
    load_frag(b3, &bsh[(3*16 + nloc)*256 + kk*32 + kb]);
    acc0 = wmma_bf16(a, b0, acc0);
    acc1 = wmma_bf16(a, b1, acc1);
    acc2 = wmma_bf16(a, b2, acc2);
    acc3 = wmma_bf16(a, b3, acc3);
  }
  #pragma unroll
  for (int t = 0; t < 4; ++t) {
    v8f acc = (t == 0) ? acc0 : (t == 1) ? acc1 : (t == 2) ? acc2 : acc3;
    int n = gn*64 + t*16 + nloc;
    float bv = bias ? bias[n] : 0.f;
    #pragma unroll
    for (int r = 0; r < 8; ++r) {
      size_t m = (size_t)tm*16 + r + (hi << 3);
      float v = acc[r] + bv;
      if (outbf) outbf[m*256 + n] = tobf(v);
      else       outf [m*256 + n] = v;
    }
  }
}

// -------- depthwise 3x3 conv on lepe_lin (SAME padding) --------
__global__ void dwconv_kernel(const __bf16* __restrict__ lin, const float* __restrict__ w,
                              const float* __restrict__ bias, __bf16* __restrict__ out){
  int i = blockIdx.x * 256 + threadIdx.x;       // B*N*C threads
  int c = i & 255; int sp = i >> 8;
  int x = sp % Ww; int yb = sp / Ww; int y = yb % Hh; int b = yb / Hh;
  float acc = bias[c];
  #pragma unroll
  for (int ky = 0; ky < 3; ++ky)
    #pragma unroll
    for (int kx = 0; kx < 3; ++kx) {
      int yy = y + ky - 1, xx = x + kx - 1;
      if (yy >= 0 && yy < Hh && xx >= 0 && xx < Ww)
        acc += tof(lin[(((size_t)b*Nn) + yy*Ww + xx)*256 + c]) * w[(ky*3 + kx)*256 + c];
    }
  out[i] = tobf(acc);
}

// -------- SR 4x4/stride4 conv as 16-tap GEMM -> xsr f32 [6272,256] --------
__global__ void sr_gemm_kernel(const __bf16* __restrict__ xb, const __bf16* __restrict__ wsrt,
                               const float* __restrict__ sr_b, float* __restrict__ xsr){
  int wid  = (blockIdx.x << 3) + (threadIdx.x >> 5);
  int lane = threadIdx.x & 31;
  int nloc = lane & 15, hi = lane >> 4, kb = hi << 3;
  int tn = wid & 15, tm = wid >> 4;
  int p = tm*16 + nloc; int b = p / N1; int pp = p % N1;
  int py = pp / 14, px = pp % 14;
  const __bf16* abase = xb + ((size_t)b*Nn + (py*4)*Ww + px*4)*256 + kb;
  const __bf16* bbase = wsrt + (size_t)(tn*16 + nloc)*4096 + kb;
  v8f acc = {};
  for (int tap = 0; tap < 16; ++tap) {
    const __bf16* arow = abase + ((tap >> 2)*Ww + (tap & 3))*256;
    const __bf16* brow = bbase + tap*256;
    #pragma unroll
    for (int kk = 0; kk < 8; ++kk) {
      Frag a, bb; load_frag(a, arow + kk*32); load_frag(bb, brow + kk*32);
      acc = wmma_bf16(a, bb, acc);
    }
  }
  int n = tn*16 + nloc;
  #pragma unroll
  for (int r = 0; r < 8; ++r) {
    size_t m = (size_t)tm*16 + r + (hi << 3);
    xsr[m*256 + n] = acc[r] + sr_b[n];
  }
}

// -------- LayerNorm + exact GELU -> x1n bf16 --------
__global__ void ln_gelu_kernel(const float* __restrict__ xsr, const float* __restrict__ g,
                               const float* __restrict__ bb, __bf16* __restrict__ out){
  __shared__ float red[256];
  int row = blockIdx.x, t = threadIdx.x;
  float v = xsr[(size_t)row*256 + t];
  red[t] = v; __syncthreads();
  for (int s = 128; s > 0; s >>= 1) { if (t < s) red[t] += red[t + s]; __syncthreads(); }
  float mean = red[0] / 256.f; __syncthreads();
  float dv = v - mean;
  red[t] = dv * dv; __syncthreads();
  for (int s = 128; s > 0; s >>= 1) { if (t < s) red[t] += red[t + s]; __syncthreads(); }
  float var = red[0] / 256.f;
  float xn = dv * rsqrtf(var + 1e-5f) * g[t] + bb[t];
  float ge = 0.5f * xn * (1.f + erff(xn * 0.70710678118654752f));
  out[(size_t)row*256 + t] = tobf(ge);
}

// -------- branch 1: global attention over 196 SR tokens (fused softmax) ----
// block = 128 threads (4 waves x 16 queries); grid = B*4*49
__global__ void attn1_kernel(const __bf16* __restrict__ q1, const __bf16* __restrict__ kv1,
                             const __bf16* __restrict__ lepe, __bf16* __restrict__ y){
  __shared__ __bf16 v1t[32 * 224];        // V transposed [ch][key], keys padded
  __shared__ __bf16 pbuf[4 * 16 * 224];   // per-wave probability tiles
  int bi = blockIdx.x;
  int chunk = bi % 49; int h = (bi / 49) & 3; int b = bi / 196;
  int t = threadIdx.x, wv = t >> 5, lane = t & 31;
  int nloc = lane & 15, hi = lane >> 4, kb = hi << 3;

  for (int idx = t; idx < 32 * 224; idx += 128) {
    int ch = idx / 224, key = idx % 224;
    __bf16 v = tobf(0.f);
    if (key < N1) v = kv1[((size_t)b*N1 + key)*256 + 128 + h*32 + ch];
    v1t[ch*224 + key] = v;
  }

  int q0 = chunk*64 + wv*16;
  Frag qa; load_frag(qa, q1 + ((size_t)b*Nn + q0 + nloc)*128 + h*32 + kb);
  const __bf16* kbase = kv1 + (size_t)b*N1*256 + h*32 + kb;
  v8f S[14];
  #pragma unroll
  for (int ct = 0; ct < 14; ++ct) {
    Frag kf; load_frag(kf, kbase + (size_t)(ct*16 + nloc)*256);
    v8f z = {}; S[ct] = wmma_bf16(qa, kf, z);
  }
  #pragma unroll
  for (int r = 0; r < 8; ++r) {
    float mx = -3.0e38f;
    #pragma unroll
    for (int ct = 0; ct < 14; ++ct) {
      float s = S[ct][r];
      s = (ct*16 + nloc < N1) ? s * SCALE : -1.0e30f;
      S[ct][r] = s; mx = fmaxf(mx, s);
    }
    for (int m = 1; m < 16; m <<= 1) mx = fmaxf(mx, __shfl_xor(mx, m, 16));
    float sum = 0.f;
    #pragma unroll
    for (int ct = 0; ct < 14; ++ct) { float e = __expf(S[ct][r] - mx); S[ct][r] = e; sum += e; }
    for (int m = 1; m < 16; m <<= 1) sum += __shfl_xor(sum, m, 16);
    float inv = 1.f / sum;
    int mr = r + (hi << 3);
    #pragma unroll
    for (int ct = 0; ct < 14; ++ct)
      pbuf[(wv*16 + mr)*224 + ct*16 + nloc] = tobf(S[ct][r] * inv);
  }
  __syncthreads();
  #pragma unroll
  for (int ct2 = 0; ct2 < 2; ++ct2) {
    v8f acc = {};
    #pragma unroll
    for (int kt = 0; kt < 7; ++kt) {
      Frag pa, vf;
      load_frag(pa, &pbuf[(wv*16 + nloc)*224 + kt*32 + kb]);
      load_frag(vf, &v1t[(ct2*16 + nloc)*224 + kt*32 + kb]);
      acc = wmma_bf16(pa, vf, acc);
    }
    #pragma unroll
    for (int r = 0; r < 8; ++r) {
      int m = r + (hi << 3);
      size_t q = (size_t)b*Nn + q0 + m;
      int ch = h*32 + ct2*16 + nloc;
      y[q*256 + ch] = tobf(acc[r] + tof(lepe[q*256 + ch]));
    }
  }
}

// -------- branch 2: 7x7 window attention (49x49, fused softmax) --------
// block = 128 threads (4 row tiles of padded 64); grid = B*4*64
__global__ void attn2_kernel(const __bf16* __restrict__ q2, const __bf16* __restrict__ kv2,
                             const __bf16* __restrict__ lepe, __bf16* __restrict__ y){
  __shared__ __bf16 v2t[32 * 64];
  __shared__ __bf16 pbuf[4 * 16 * 64];
  int bi = blockIdx.x;
  int w = bi & 63, h = (bi >> 6) & 3, b = bi >> 8;
  int wy = w >> 3, wx = w & 7;
  int t = threadIdx.x, wv = t >> 5, lane = t & 31;
  int nloc = lane & 15, hi = lane >> 4, kb = hi << 3;
  auto widx = [&](int i){ return (wy*7 + i/7)*Ww + wx*7 + (i%7); };

  for (int idx = t; idx < 32 * 64; idx += 128) {
    int ch = idx >> 6, key = idx & 63;
    __bf16 v = tobf(0.f);
    if (key < 49) v = kv2[((size_t)b*Nn + widx(key))*256 + 128 + h*32 + ch];
    v2t[(ch << 6) + key] = v;
  }

  int qi = wv*16 + nloc; int qic = (qi < 49) ? qi : 0;
  Frag qa; load_frag(qa, q2 + ((size_t)b*Nn + widx(qic))*128 + h*32 + kb);
  v8f S[4];
  #pragma unroll
  for (int ct = 0; ct < 4; ++ct) {
    int kix = ct*16 + nloc; int kic = (kix < 49) ? kix : 0;
    Frag kf; load_frag(kf, kv2 + ((size_t)b*Nn + widx(kic))*256 + h*32 + kb);
    v8f z = {}; S[ct] = wmma_bf16(qa, kf, z);
  }
  #pragma unroll
  for (int r = 0; r < 8; ++r) {
    float mx = -3.0e38f;
    #pragma unroll
    for (int ct = 0; ct < 4; ++ct) {
      float s = S[ct][r];
      s = (ct*16 + nloc < 49) ? s * SCALE : -1.0e30f;
      S[ct][r] = s; mx = fmaxf(mx, s);
    }
    for (int m = 1; m < 16; m <<= 1) mx = fmaxf(mx, __shfl_xor(mx, m, 16));
    float sum = 0.f;
    #pragma unroll
    for (int ct = 0; ct < 4; ++ct) { float e = __expf(S[ct][r] - mx); S[ct][r] = e; sum += e; }
    for (int m = 1; m < 16; m <<= 1) sum += __shfl_xor(sum, m, 16);
    float inv = 1.f / sum;
    int mr = r + (hi << 3);
    #pragma unroll
    for (int ct = 0; ct < 4; ++ct)
      pbuf[(wv*16 + mr)*64 + ct*16 + nloc] = tobf(S[ct][r] * inv);
  }
  __syncthreads();
  #pragma unroll
  for (int ct2 = 0; ct2 < 2; ++ct2) {
    v8f acc = {};
    #pragma unroll
    for (int kt = 0; kt < 2; ++kt) {
      Frag pa, vf;
      load_frag(pa, &pbuf[(wv*16 + nloc)*64 + kt*32 + kb]);
      load_frag(vf, &v2t[(ct2*16 + nloc)*64 + kt*32 + kb]);
      acc = wmma_bf16(pa, vf, acc);
    }
    #pragma unroll
    for (int r = 0; r < 8; ++r) {
      int m = r + (hi << 3);
      int qr = wv*16 + m;
      if (qr < 49) {
        size_t q = (size_t)b*Nn + widx(qr);
        int ch = 128 + h*32 + ct2*16 + nloc;
        y[q*256 + ch] = tobf(acc[r] + tof(lepe[q*256 + ch]));
      }
    }
  }
}

// ---------------- host side ----------------
extern "C" void kernel_launch(void* const* d_in, const int* in_sizes, int n_in,
                              void* d_out, int out_size, void* d_ws, size_t ws_size,
                              hipStream_t stream) {
  const float* x          = (const float*)d_in[0];
  const float* lepe_lin_w = (const float*)d_in[1];
  const float* lepe_lin_b = (const float*)d_in[2];
  const float* lepe_conv_w= (const float*)d_in[3];
  const float* lepe_conv_b= (const float*)d_in[4];
  const float* sr_w       = (const float*)d_in[5];
  const float* sr_b       = (const float*)d_in[6];
  const float* norm_g     = (const float*)d_in[7];
  const float* norm_b     = (const float*)d_in[8];
  const float* q1_w       = (const float*)d_in[9];
  const float* kv1_w      = (const float*)d_in[10];
  const float* q2_w       = (const float*)d_in[11];
  const float* kv2_w      = (const float*)d_in[12];
  const float* proj_w     = (const float*)d_in[13];
  const float* proj_b     = (const float*)d_in[14];

  char* base = (char*)d_ws;
  size_t off = 0;
  auto alloc = [&](size_t bytes){ char* p = base + off; off = (off + bytes + 255) & ~(size_t)255; return p; };

  __bf16* xb    = (__bf16*)alloc((size_t)Mrows*256*2);
  __bf16* wcat  = (__bf16*)alloc(768*256*2);
  __bf16* wkv1t = (__bf16*)alloc(256*256*2);
  __bf16* wprojt= (__bf16*)alloc(256*256*2);
  __bf16* wsrt  = (__bf16*)alloc(256*16*256*2);
  __bf16* lin   = (__bf16*)alloc((size_t)Mrows*256*2);
  __bf16* lepe  = (__bf16*)alloc((size_t)Mrows*256*2);
  __bf16* q1    = (__bf16*)alloc((size_t)Mrows*128*2);
  __bf16* q2    = (__bf16*)alloc((size_t)Mrows*128*2);
  __bf16* kv2   = (__bf16*)alloc((size_t)Mrows*256*2);
  float*  xsr   = (float*) alloc((size_t)P1*256*4);
  __bf16* x1n   = (__bf16*)alloc((size_t)P1*256*2);
  __bf16* kv1   = (__bf16*)alloc((size_t)KV1R*256*2);
  __bf16* ybuf  = (__bf16*)alloc((size_t)Mrows*256*2);
  (void)ws_size; (void)n_in; (void)in_sizes; (void)out_size;

  // 1. x -> bf16
  cvt_bf16_kernel<<<Mrows, 256, 0, stream>>>(x, xb, Mrows*256);
  // 2. weight prep (concat/transpose to N-major bf16)
  prep_weights_kernel<<<5376, 256, 0, stream>>>(lepe_lin_w, q1_w, q2_w, kv2_w,
                                                kv1_w, proj_w, sr_w,
                                                wcat, wkv1t, wprojt, wsrt);
  // 3. fused projections: lepe_lin | q1 | q2 | kv2  (784 M-blocks x 12 N-groups)
  gemm_fused_kernel<<<784*12, 256, 0, stream>>>(xb, wcat, lepe_lin_b, lin, q1, q2, kv2);
  // 4. depthwise 3x3 -> lepe
  dwconv_kernel<<<Mrows, 256, 0, stream>>>(lin, lepe_conv_w, lepe_conv_b, lepe);
  // 5. SR conv as GEMM (392*16 tiles)
  sr_gemm_kernel<<<784, 256, 0, stream>>>(xb, wsrt, sr_b, xsr);
  // 6. LayerNorm + GELU
  ln_gelu_kernel<<<P1, 256, 0, stream>>>(xsr, norm_g, norm_b, x1n);
  // 7. kv1 GEMM (49 M-blocks x 4 N-groups)
  gemm_k256_kernel<<<49*4, 256, 0, stream>>>(x1n, wkv1t, nullptr, kv1, nullptr);
  // 8. branch-1 attention (+lepe) -> y[:, :128]
  attn1_kernel<<<Bc*H2*49, 128, 0, stream>>>(q1, kv1, lepe, ybuf);
  // 9. branch-2 window attention (+lepe) -> y[:, 128:]
  attn2_kernel<<<Bc*H2*64, 128, 0, stream>>>(q2, kv2, lepe, ybuf);
  // 10. output projection (784 M-blocks x 4 N-groups) -> d_out f32
  gemm_k256_kernel<<<784*4, 256, 0, stream>>>(ybuf, wprojt, proj_b, nullptr, (float*)d_out);
}
